// PhaseSpaceIntegrator_42640435314812
// MI455X (gfx1250) — matrix operand order repaired and checked by reading, not compiled, not run on hardware
//
#include <hip/hip_runtime.h>
#include <hip/hip_bf16.h>
#include <math.h>

// Problem constants (match reference)
#define V_   32000
#define D_   512
#define L_   6
#define B_   2
#define S_   2048
#define BS_  (B_ * S_)          // 4096 rows
#define BSD_ ((size_t)BS_ * D_) // 2,097,152
#define CH_  64                 // scan chunks
#define CS_  (S_ / CH_)         // 32 steps per chunk

typedef __attribute__((ext_vector_type(16))) _Float16 v16h;
typedef __attribute__((ext_vector_type(8)))  _Float16 v8h;
typedef __attribute__((ext_vector_type(8)))  float    v8f;

#define LDST 40   // LDS row stride in f16 (32 data + 8 pad; 80B, 16B aligned)

#if defined(__gfx1250__) && __has_builtin(__builtin_amdgcn_global_load_async_to_lds_b128)
#define USE_ASYNC_LDS 1
typedef int i32x4_ __attribute__((vector_size(16)));
typedef __attribute__((address_space(1))) i32x4_* gas_p;   // global b128 ptr
typedef __attribute__((address_space(3))) i32x4_* las_p;   // LDS b128 ptr
#else
#define USE_ASYNC_LDS 0
#endif

// ---------------------------------------------------------------------------
// Embedding: h[b,s,:] = emb[x[b,s],:] + pos_enc[s,:]
// ---------------------------------------------------------------------------
__global__ __launch_bounds__(256) void embed_kernel(
    const int* __restrict__ x, const float* __restrict__ emb,
    const float* __restrict__ pos, float* __restrict__ h) {
  size_t i = (size_t)blockIdx.x * 256 + threadIdx.x;   // over BSD_
  int d = (int)(i & (D_ - 1));
  size_t row = i >> 9;                                  // D_=512 = 2^9
  int s = (int)(row & (S_ - 1));
  int tok = x[row];
  h[i] = emb[(size_t)tok * D_ + d] + pos[(size_t)s * D_ + d];
}

// ---------------------------------------------------------------------------
// f32 -> f16 elementwise (emb for tied head)
// ---------------------------------------------------------------------------
__global__ __launch_bounds__(256) void cvt_f16_kernel(
    const float* __restrict__ in, _Float16* __restrict__ out, size_t n) {
  size_t i = (size_t)blockIdx.x * 256 + threadIdx.x;
  if (i < n) out[i] = (_Float16)in[i];
}

// ---------------------------------------------------------------------------
// Per-layer weight transpose+convert: src f32 [K][N] -> dst f16 [N][K]
// blockIdx.z selects the layer (stride K*N in both).
// ---------------------------------------------------------------------------
__global__ __launch_bounds__(256) void transpose_f16_kernel(
    const float* __restrict__ src, _Float16* __restrict__ dst, int K, int N) {
  __shared__ float tile[32][33];
  size_t lo = (size_t)blockIdx.z * K * N;
  src += lo; dst += lo;
  int n0 = blockIdx.x * 32, k0 = blockIdx.y * 32;
  int tx = threadIdx.x & 31, ty = threadIdx.x >> 5;     // 32 x 8
  #pragma unroll
  for (int r = ty; r < 32; r += 8)
    tile[r][tx] = src[(size_t)(k0 + r) * N + n0 + tx];
  __syncthreads();
  #pragma unroll
  for (int r = ty; r < 32; r += 8)
    dst[(size_t)(n0 + r) * K + k0 + tx] = (_Float16)tile[tx][r];
}

// ---------------------------------------------------------------------------
// Row LayerNorm over RL columns; optional f32 and f16 outputs.
// One row per 256-thread block.
// ---------------------------------------------------------------------------
template <int RL, bool W32, bool W16>
__global__ __launch_bounds__(256) void ln_kernel(
    const float* __restrict__ in, const float* __restrict__ g,
    const float* __restrict__ b, float* __restrict__ out32,
    _Float16* __restrict__ out16) {
  constexpr int T = 256;
  constexpr int PER = RL / T;
  size_t row = blockIdx.x;
  const float* xr = in + row * RL;
  float v[PER];
  float s = 0.f, s2 = 0.f;
  #pragma unroll
  for (int i = 0; i < PER; ++i) {
    v[i] = xr[threadIdx.x + i * T];
    s += v[i];
    s2 += v[i] * v[i];
  }
  #pragma unroll
  for (int off = 16; off > 0; off >>= 1) {
    s  += __shfl_xor(s,  off, 32);
    s2 += __shfl_xor(s2, off, 32);
  }
  __shared__ float red1[8], red2[8];
  int wave = threadIdx.x >> 5, lane = threadIdx.x & 31;
  if (lane == 0) { red1[wave] = s; red2[wave] = s2; }
  __syncthreads();
  if (wave == 0) {
    s  = (lane < 8) ? red1[lane] : 0.f;
    s2 = (lane < 8) ? red2[lane] : 0.f;
    #pragma unroll
    for (int off = 4; off > 0; off >>= 1) {
      s  += __shfl_xor(s,  off, 32);
      s2 += __shfl_xor(s2, off, 32);
    }
    if (lane == 0) { red1[0] = s; red2[0] = s2; }
  }
  __syncthreads();
  float mean = red1[0] * (1.0f / RL);
  float var  = red2[0] * (1.0f / RL) - mean * mean;
  float r = rsqrtf(var + 1e-5f);
  #pragma unroll
  for (int i = 0; i < PER; ++i) {
    int c = threadIdx.x + i * T;
    float y = (v[i] - mean) * r * g[c] + b[c];
    if (W32) out32[row * RL + c] = y;
    if (W16) out16[row * RL + c] = (_Float16)y;
  }
}

// ---------------------------------------------------------------------------
// Chunked phase-space scan (3-level linear scan).
// Pass A: per-chunk partial sums of the phase increments.
// ---------------------------------------------------------------------------
__global__ __launch_bounds__(512) void scan_phase_partial_kernel(
    const float* __restrict__ omega, const float* __restrict__ log_scale,
    float* __restrict__ P) {
  int b = blockIdx.x, c = blockIdx.y, d = threadIdx.x;
  float esc = expf(log_scale[d]);
  const float* po = omega + ((size_t)b * S_ + (size_t)c * CS_) * D_ + d;
  float acc = 0.f;
  #pragma unroll 4
  for (int s = 0; s < CS_; ++s) {
    float posv = (float)(c * CS_ + s + 1);
    acc += po[(size_t)s * D_] * esc * rsqrtf(posv);
  }
  P[((size_t)b * CH_ + c) * D_ + d] = acc;
}

// In-place exclusive prefix over the chunk axis: A is [B][CH][D].
__global__ __launch_bounds__(512) void scan_prefix_kernel(float* __restrict__ A) {
  int b = blockIdx.x, d = threadIdx.x;
  float run = 0.f;
  for (int c = 0; c < CH_; ++c) {
    size_t idx = ((size_t)b * CH_ + c) * D_ + d;
    float t = A[idx];
    A[idx] = run;
    run += t;
  }
}

// Pass C: with exclusive phase prefix, compute phi/cos/sin, zc/zs (written to
// ctx cols 0/1), stash c,s, and per-chunk partial sums of zc/zs.
__global__ __launch_bounds__(512) void scan_mid_kernel(
    const float* __restrict__ omega, const float* __restrict__ z,
    const float* __restrict__ log_scale, const float* __restrict__ P,
    float* __restrict__ ctx, float* __restrict__ cbuf, float* __restrict__ sbuf,
    float* __restrict__ MR, float* __restrict__ MI) {
  int b = blockIdx.x, c = blockIdx.y, d = threadIdx.x;
  float esc = expf(log_scale[d]);
  size_t base = ((size_t)b * S_ + (size_t)c * CS_) * D_ + d;
  const float* po = omega + base;
  const float* pz = z + base;
  float* pcb = cbuf + base;
  float* psb = sbuf + base;
  float* pc = ctx + ((size_t)b * S_ + (size_t)c * CS_) * (4 * D_) + d;
  float phi = P[((size_t)b * CH_ + c) * D_ + d];
  float mr = 0.f, mi = 0.f;
  #pragma unroll 2
  for (int s = 0; s < CS_; ++s) {
    float posv = (float)(c * CS_ + s + 1);
    phi += po[(size_t)s * D_] * esc * rsqrtf(posv);
    float sn, cs;
    __sincosf(phi, &sn, &cs);
    float zv = pz[(size_t)s * D_];
    float zc = zv * cs, zs = zv * sn;
    mr += zc; mi += zs;
    pcb[(size_t)s * D_] = cs;
    psb[(size_t)s * D_] = sn;
    float* o = pc + (size_t)s * (4 * D_);
    o[0]  = zc;
    o[D_] = zs;
  }
  MR[((size_t)b * CH_ + c) * D_ + d] = mr;
  MI[((size_t)b * CH_ + c) * D_ + d] = mi;
}

// Pass E: with exclusive zc/zs prefixes, produce rr/ri (ctx cols 2/3).
__global__ __launch_bounds__(512) void scan_final_kernel(
    const float* __restrict__ MR, const float* __restrict__ MI,
    const float* __restrict__ cbuf, const float* __restrict__ sbuf,
    float* __restrict__ ctx) {
  int b = blockIdx.x, c = blockIdx.y, d = threadIdx.x;
  size_t base = ((size_t)b * S_ + (size_t)c * CS_) * D_ + d;
  const float* pcb = cbuf + base;
  const float* psb = sbuf + base;
  float* pc = ctx + ((size_t)b * S_ + (size_t)c * CS_) * (4 * D_) + d;
  float mr = MR[((size_t)b * CH_ + c) * D_ + d];
  float mi = MI[((size_t)b * CH_ + c) * D_ + d];
  #pragma unroll 2
  for (int s = 0; s < CS_; ++s) {
    float* o = pc + (size_t)s * (4 * D_);
    float zc = o[0];
    float zs = o[D_];
    mr += zc; mi += zs;
    float inv = 1.0f / (float)(c * CS_ + s + 1);
    float cs = pcb[(size_t)s * D_];
    float sn = psb[(size_t)s * D_];
    float mrn = mr * inv, minr = mi * inv;
    o[2 * D_] = mrn * cs + minr * sn;
    o[3 * D_] = minr * cs - mrn * sn;
  }
}

// ---------------------------------------------------------------------------
// WMMA GEMM: C[M,N] = A[M,K] @ B^T  where Bw is f16 [N][K] (K contiguous).
// 256 threads / 8 waves; block tile 128x128; wave tile 32x64; K-step 32;
// double-buffered LDS, async global->LDS when available.
// MODE 0: Cf = acc + bias                       (f32 out)
// MODE 1: Ch = gelu(acc + bias)                 (f16 out, exact erf GELU)
// MODE 2: Hres += Zres + (acc + bias)           (residual update, f32)
// MODE 3: Cf = acc                              (no bias; logits)
// ---------------------------------------------------------------------------
template <int MODE>
__global__ __launch_bounds__(256) void gemm_wmma_kernel(
    const _Float16* __restrict__ A, const _Float16* __restrict__ Bw,
    const float* __restrict__ bias, float* __restrict__ Cf,
    _Float16* __restrict__ Ch, const float* __restrict__ Zres,
    float* __restrict__ Hres, int N, int K) {
  __shared__ __align__(16) _Float16 As[2][128 * LDST];
  __shared__ __align__(16) _Float16 Bs[2][128 * LDST];

  int tid  = threadIdx.x;
  int wave = tid >> 5, lane = tid & 31;
  int wm = wave & 3, wn = wave >> 2;          // 4x2 wave grid -> 128x128 tile
  int bm = blockIdx.y * 128, bn = blockIdx.x * 128;
  int n16 = lane & 15, lhi = lane >> 4;

  // cooperative-load coordinates: 16B chunks, 4 per thread (2 A + 2 B)
  int srow = tid >> 2, scol = (tid & 3) << 3;

  const _Float16* Ab = A  + (size_t)bm * K + (size_t)srow * K + scol;
  const _Float16* Bb = Bw + (size_t)bn * K + (size_t)srow * K + scol;
  size_t rstride = (size_t)64 * K;   // 64 rows down

  auto stage = [&](int buf, int k0) {
    _Float16* la0 = &As[buf][srow * LDST + scol];
    _Float16* la1 = &As[buf][(srow + 64) * LDST + scol];
    _Float16* lb0 = &Bs[buf][srow * LDST + scol];
    _Float16* lb1 = &Bs[buf][(srow + 64) * LDST + scol];
#if USE_ASYNC_LDS
    __builtin_amdgcn_global_load_async_to_lds_b128(
        (gas_p)(Ab + k0), (las_p)la0, 0, 0);
    __builtin_amdgcn_global_load_async_to_lds_b128(
        (gas_p)(Ab + rstride + k0), (las_p)la1, 0, 0);
    __builtin_amdgcn_global_load_async_to_lds_b128(
        (gas_p)(Bb + k0), (las_p)lb0, 0, 0);
    __builtin_amdgcn_global_load_async_to_lds_b128(
        (gas_p)(Bb + rstride + k0), (las_p)lb1, 0, 0);
#else
    v8h a0 = *(const v8h*)(Ab + k0);
    v8h a1 = *(const v8h*)(Ab + rstride + k0);
    v8h b0 = *(const v8h*)(Bb + k0);
    v8h b1 = *(const v8h*)(Bb + rstride + k0);
    *(v8h*)la0 = a0;
    *(v8h*)la1 = a1;
    *(v8h*)lb0 = b0;
    *(v8h*)lb1 = b1;
    __builtin_prefetch(Ab + k0 + 32, 0, 1);
    __builtin_prefetch(Bb + k0 + 32, 0, 1);
#endif
  };

  v8f acc[2][4] = {};

  auto compute = [&](int buf) {
    v16h af[2], bf[4];
    #pragma unroll
    for (int mt = 0; mt < 2; ++mt) {
      // A layout: elems 0-7 -> K = 8*lhi + 0..7 ; elems 8-15 -> K = 16+8*lhi+0..7
      const _Float16* p = &As[buf][(wm * 32 + mt * 16 + n16) * LDST];
      v8h lo = *(const v8h*)(p + 8 * lhi);
      v8h hi = *(const v8h*)(p + 16 + 8 * lhi);
      af[mt] = __builtin_shufflevector(lo, hi, 0, 1, 2, 3, 4, 5, 6, 7,
                                       8, 9, 10, 11, 12, 13, 14, 15);
    }
    #pragma unroll
    for (int nt = 0; nt < 4; ++nt) {
      // B layout: elems 0-7 -> K = 8*lhi + 0..7 ; elems 8-15 -> K = 16+8*lhi+0..7
      const _Float16* p = &Bs[buf][(wn * 64 + nt * 16 + n16) * LDST];
      v8h lo = *(const v8h*)(p + 8 * lhi);
      v8h hi = *(const v8h*)(p + 16 + 8 * lhi);
      bf[nt] = __builtin_shufflevector(lo, hi, 0, 1, 2, 3, 4, 5, 6, 7,
                                       8, 9, 10, 11, 12, 13, 14, 15);
    }
    #pragma unroll
    for (int mt = 0; mt < 2; ++mt)
      #pragma unroll
      for (int nt = 0; nt < 4; ++nt)
        acc[mt][nt] = __builtin_amdgcn_wmma_f32_16x16x32_f16(
            false, af[mt], false, bf[nt], (short)0, acc[mt][nt], false, false);
  };

  stage(0, 0);
#if USE_ASYNC_LDS
  asm volatile("s_wait_asynccnt 0" ::: "memory");
#endif
  __syncthreads();

  int buf = 0;
  for (int k0 = 32; k0 < K; k0 += 32) {
    stage(buf ^ 1, k0);
    compute(buf);
#if USE_ASYNC_LDS
    asm volatile("s_wait_asynccnt 0" ::: "memory");
#endif
    __syncthreads();
    buf ^= 1;
  }
  compute(buf);

  // ---- epilogue ----
  #pragma unroll
  for (int mt = 0; mt < 2; ++mt) {
    #pragma unroll
    for (int nt = 0; nt < 4; ++nt) {
      int col = bn + wn * 64 + nt * 16 + n16;
      float bv = (MODE == 3) ? 0.f : bias[col];
      #pragma unroll
      for (int v = 0; v < 8; ++v) {
        int rowg = bm + wm * 32 + mt * 16 + v + 8 * lhi;
        size_t idx = (size_t)rowg * N + col;
        float val = acc[mt][nt][v] + bv;
        if (MODE == 0) {
          Cf[idx] = val;
        } else if (MODE == 1) {
          float ge = 0.5f * val * (1.0f + erff(val * 0.70710678118f));
          Ch[idx] = (_Float16)ge;
        } else if (MODE == 2) {
          Hres[idx] = Hres[idx] + Zres[idx] + val;
        } else {
          Cf[idx] = val;
        }
      }
    }
  }
}

// ---------------------------------------------------------------------------
extern "C" void kernel_launch(void* const* d_in, const int* in_sizes, int n_in,
                              void* d_out, int out_size, void* d_ws, size_t ws_size,
                              hipStream_t stream) {
  (void)in_sizes; (void)n_in; (void)out_size; (void)ws_size;
  const int*   x         = (const int*)  d_in[0];
  const float* emb       = (const float*)d_in[1];
  const float* pos_enc   = (const float*)d_in[2];
  const float* norm_g    = (const float*)d_in[3];
  const float* norm_b    = (const float*)d_in[4];
  const float* omega_W   = (const float*)d_in[5];
  const float* omega_b   = (const float*)d_in[6];
  const float* log_scale = (const float*)d_in[7];
  const float* ln4_g     = (const float*)d_in[8];
  const float* ln4_b     = (const float*)d_in[9];
  const float* w1        = (const float*)d_in[10];
  const float* b1        = (const float*)d_in[11];
  const float* w2        = (const float*)d_in[12];
  const float* b2        = (const float*)d_in[13];
  const float* fin_g     = (const float*)d_in[14];
  const float* fin_b     = (const float*)d_in[15];
  float* out = (float*)d_out;

  // workspace carve-up
  char* p = (char*)d_ws;
  auto alloc = [&](size_t bytes) -> void* {
    void* r = (void*)p;
    p += (bytes + 255) & ~(size_t)255;
    return r;
  };
  float*    h      = (float*)   alloc(BSD_ * 4);                 // [BS, D]
  float*    z      = (float*)   alloc(BSD_ * 4);                 // [BS, D]
  _Float16* z16    = (_Float16*)alloc(BSD_ * 2);
  float*    omg    = (float*)   alloc(BSD_ * 4);                 // omega [BS, D]
  float*    ctx    = (float*)   alloc(BSD_ * 4 * 4);             // [BS, 4D] f32
  _Float16* ctx16  = (_Float16*)alloc(BSD_ * 4 * 2);             // [BS, 4D] f16
  _Float16* hid16  = (_Float16*)alloc(BSD_ * 2 * 2);             // [BS, 2D] f16
  _Float16* hfin16 = (_Float16*)alloc(BSD_ * 2);
  float*    cbuf   = (float*)   alloc(BSD_ * 4);                 // cos(phi)
  float*    sbuf   = (float*)   alloc(BSD_ * 4);                 // sin(phi)
  float*    Pp     = (float*)   alloc((size_t)B_ * CH_ * D_ * 4);
  float*    MRp    = (float*)   alloc((size_t)B_ * CH_ * D_ * 4);
  float*    MIp    = (float*)   alloc((size_t)B_ * CH_ * D_ * 4);
  _Float16* emb16  = (_Float16*)alloc((size_t)V_ * D_ * 2);      // [V, D]
  _Float16* wOT    = (_Float16*)alloc((size_t)L_ * D_ * D_ * 2);         // [L][D][D]
  _Float16* w1T    = (_Float16*)alloc((size_t)L_ * 2 * D_ * 4 * D_ * 2); // [L][2D][4D]
  _Float16* w2T    = (_Float16*)alloc((size_t)L_ * D_ * 2 * D_ * 2);     // [L][D][2D]

  // ---- one-time per launch: weight conversion ----
  transpose_f16_kernel<<<dim3(D_ / 32, D_ / 32, L_), 256, 0, stream>>>(
      omega_W, wOT, D_, D_);
  transpose_f16_kernel<<<dim3(2 * D_ / 32, 4 * D_ / 32, L_), 256, 0, stream>>>(
      w1, w1T, 4 * D_, 2 * D_);
  transpose_f16_kernel<<<dim3(D_ / 32, 2 * D_ / 32, L_), 256, 0, stream>>>(
      w2, w2T, 2 * D_, D_);
  {
    size_t n = (size_t)V_ * D_;
    cvt_f16_kernel<<<(unsigned)((n + 255) / 256), 256, 0, stream>>>(emb, emb16, n);
  }

  // ---- embed ----
  embed_kernel<<<(unsigned)(BSD_ / 256), 256, 0, stream>>>(x, emb, pos_enc, h);

  // ---- layers ----
  for (int l = 0; l < L_; ++l) {
    ln_kernel<D_, true, true><<<BS_, 256, 0, stream>>>(
        h, norm_g + (size_t)l * D_, norm_b + (size_t)l * D_, z, z16);

    gemm_wmma_kernel<0><<<dim3(D_ / 128, BS_ / 128), 256, 0, stream>>>(
        z16, wOT + (size_t)l * D_ * D_, omega_b + (size_t)l * D_,
        omg, nullptr, nullptr, nullptr, D_, D_);

    // chunk-parallel phase-space scan
    scan_phase_partial_kernel<<<dim3(B_, CH_), D_, 0, stream>>>(
        omg, log_scale + (size_t)l * D_, Pp);
    scan_prefix_kernel<<<B_, D_, 0, stream>>>(Pp);
    scan_mid_kernel<<<dim3(B_, CH_), D_, 0, stream>>>(
        omg, z, log_scale + (size_t)l * D_, Pp, ctx, cbuf, sbuf, MRp, MIp);
    scan_prefix_kernel<<<B_, D_, 0, stream>>>(MRp);
    scan_prefix_kernel<<<B_, D_, 0, stream>>>(MIp);
    scan_final_kernel<<<dim3(B_, CH_), D_, 0, stream>>>(
        MRp, MIp, cbuf, sbuf, ctx);

    ln_kernel<4 * D_, false, true><<<BS_, 256, 0, stream>>>(
        ctx, ln4_g + (size_t)l * 4 * D_, ln4_b + (size_t)l * 4 * D_,
        nullptr, ctx16);

    gemm_wmma_kernel<1><<<dim3(2 * D_ / 128, BS_ / 128), 256, 0, stream>>>(
        ctx16, w1T + (size_t)l * 2 * D_ * 4 * D_, b1 + (size_t)l * 2 * D_,
        nullptr, hid16, nullptr, nullptr, 2 * D_, 4 * D_);

    gemm_wmma_kernel<2><<<dim3(D_ / 128, BS_ / 128), 256, 0, stream>>>(
        hid16, w2T + (size_t)l * D_ * 2 * D_, b2 + (size_t)l * D_,
        nullptr, nullptr, z, h, D_, 2 * D_);
  }

  // ---- final LN + tied head ----
  ln_kernel<D_, false, true><<<BS_, 256, 0, stream>>>(
      h, fin_g, fin_b, nullptr, hfin16);

  gemm_wmma_kernel<3><<<dim3(V_ / 128, BS_ / 128), 256, 0, stream>>>(
      hfin16, emb16, nullptr, out, nullptr, nullptr, nullptr, V_, D_);
}
